// CVKANLayer_30751965839657
// MI455X (gfx1250) — matrix operand order repaired and checked by reading, not compiled, not run on hardware
//
#include <hip/hip_runtime.h>
#include <stdint.h>

// CDNA5 / gfx1250 fused CV-KAN layer, software-pipelined, double-buffered LDS.
// out[b,o,{re,im}] = sum_{i,u,v} eu(b,i,u)*ev(b,i,v)*W{r,c}[i,o,u,v]
//                  + [s_re|s_im] @ [Wre;-Wim] (re) / [Wim;Wre] (im) + sum_i bias
// bf16 WMMA (v_wmma_f32_16x16x32_bf16), f32 accumulate.
// bf16 conversion via native casts -> v_cvt_pk_bf16_f32 instead of manual RNE.

typedef __attribute__((ext_vector_type(16))) __bf16 bf16x16;
typedef __attribute__((ext_vector_type(2)))  __bf16 bf16x2;
typedef __attribute__((ext_vector_type(8)))  float  f32x8;

struct U4 { unsigned x, y, z, w; };
union FragCast { U4 q[2]; bf16x16 v; };

__device__ __forceinline__ unsigned short f2bf(float f) {
  return __builtin_bit_cast(unsigned short, (__bf16)f);
}
__device__ __forceinline__ unsigned pack2(float a, float b) {
  bf16x2 v;
  v.x = (__bf16)a;
  v.y = (__bf16)b;
  return __builtin_bit_cast(unsigned, v);
}

#define TM  128      // block rows (batch)
#define TN  64       // block cols (output channels)
#define LDK 72       // padded k stride in bf16 elements (36 dwords -> bank spread)

// dynamic-LDS layout (in ushort units)
#define A_OFF(buf)   ((buf) * (TM * LDK))                  // 2 x 9216
#define BR_OFF(buf)  (2 * TM * LDK + (buf) * (TN * LDK))   // 2 x 4608
#define BI_OFF(buf)  (2 * TM * LDK + 2 * TN * LDK + (buf) * (TN * LDK))
#define BIAS_BYTE    (2 * (TM + 2 * TN) * LDK * 2)         // 73728
#define SMEM_BYTES   (BIAS_BYTE + 2 * TN * 4)              // 74240

struct MainRegs {
  float  xr, xi;                       // A-tile inputs
  float4 r0, r1, r2, r3;               // realweights row chunk
  float4 c0, c1, c2, c3;               // complexweights row chunk
};

__global__ __launch_bounds__(256)
void cvkan_fused(const float* __restrict__ x_re, const float* __restrict__ x_im,
                 const float* __restrict__ wR,   const float* __restrict__ wC,
                 const float* __restrict__ swr,  const float* __restrict__ swi,
                 const float* __restrict__ sbr,  const float* __restrict__ sbi,
                 float* __restrict__ out)
{
  extern __shared__ __align__(16) unsigned char smem_raw[];
  unsigned short* SM = (unsigned short*)smem_raw;
  float* biasR = (float*)(smem_raw + BIAS_BYTE);
  float* biasI = biasR + TN;

  const int tid  = threadIdx.x;
  const int lane = tid & 31;          // wave32
  const int wave = tid >> 5;          // 0..7
  const int wm   = wave >> 1;         // 0..3  (M subtile)
  const int wn   = wave & 1;          // 0..1  (N subtile)
  const int m_base = blockIdx.x * TM;
  const int n_base = blockIdx.y * TN;

  // ---- bias column sums (sum over i of silu_bias[i,o]) ----
  if (tid < 64) {
    float s = 0.0f;
    const float* p = sbr + (n_base + tid);
    for (int i = 0; i < 256; ++i) s += p[i * 256];
    biasR[tid] = s;
  } else if (tid < 128) {
    const int o = tid - 64;
    float s = 0.0f;
    const float* p = sbi + (n_base + o);
    for (int i = 0; i < 256; ++i) s += p[i * 256];
    biasI[o] = s;
  }

  f32x8 accR[2][2], accI[2][2];
#pragma unroll
  for (int mi = 0; mi < 2; ++mi)
#pragma unroll
    for (int ni = 0; ni < 2; ++ni)
#pragma unroll
      for (int k = 0; k < 8; ++k) { accR[mi][ni][k] = 0.0f; accI[mi][ni][k] = 0.0f; }

  // producer-thread roles
  const int ar = tid >> 1;   // 0..127 : A-tile row
  const int ah = tid & 1;    // 0/1    : A-tile k half (32 bf16 each)
  const int bo = tid >> 2;   // 0..63  : B-tile output channel
  const int bq = tid & 3;    // 0..3   : B-tile k quarter (16 bf16 each)

  const int lr = lane & 15;  // WMMA fragment row/col within 16
  const int kh = lane >> 4;  // WMMA fragment k-half selector

  const float step = 4.0f / 7.0f;     // linspace(-2, 2, 8)

  // ---- stage 1: global -> registers (issued ahead of the WMMA burst) ----
  auto load_main = [&](int i, MainRegs& m) {
    m.xr = x_re[(size_t)(m_base + ar) * 256 + i];
    m.xi = x_im[(size_t)(m_base + ar) * 256 + i];
    const size_t rowoff = ((size_t)i * 256 + (size_t)(n_base + bo)) * 64 + (size_t)bq * 16;
    const float4* pr = (const float4*)(wR + rowoff);
    const float4* pc = (const float4*)(wC + rowoff);
    m.r0 = pr[0]; m.r1 = pr[1]; m.r2 = pr[2]; m.r3 = pr[3];
    m.c0 = pc[0]; m.c1 = pc[1]; m.c2 = pc[2]; m.c3 = pc[3];
    if (i + 1 < 256) {                 // warm L2/L0 for the step after
      __builtin_prefetch(wR + rowoff + 256 * 64, 0, 0);
      __builtin_prefetch(wC + rowoff + 256 * 64, 0, 0);
    }
  };

  // ---- stage 2: convert + LDS store into buffer `buf` ----
  auto store_main = [&](const MainRegs& m, int buf) {
    // A[r][u*8+v] = exp(-(xr-lin[u])^2) * exp(-(xi-lin[v])^2)
    float ev[8];
#pragma unroll
    for (int v = 0; v < 8; ++v) {
      float d = m.xi - (-2.0f + (float)v * step);
      ev[v] = __expf(-d * d);
    }
    U4* arow = (U4*)(SM + A_OFF(buf)) + (ar * 9 + ah * 4);   // 36 dwords = 9 U4 per row
#pragma unroll
    for (int uu = 0; uu < 4; ++uu) {
      float du = m.xr - (-2.0f + (float)(ah * 4 + uu) * step);
      float eu = __expf(-du * du);
      U4 p;
      p.x = pack2(eu * ev[0], eu * ev[1]);
      p.y = pack2(eu * ev[2], eu * ev[3]);
      p.z = pack2(eu * ev[4], eu * ev[5]);
      p.w = pack2(eu * ev[6], eu * ev[7]);
      arow[uu] = p;
    }
    // B tiles: [o][k] bf16, k contiguous
    U4* br = (U4*)(SM + BR_OFF(buf)) + (bo * 9 + bq * 2);
    U4* bi = (U4*)(SM + BI_OFF(buf)) + (bo * 9 + bq * 2);
    U4 p;
    p.x = pack2(m.r0.x, m.r0.y); p.y = pack2(m.r0.z, m.r0.w);
    p.z = pack2(m.r1.x, m.r1.y); p.w = pack2(m.r1.z, m.r1.w);
    br[0] = p;
    p.x = pack2(m.r2.x, m.r2.y); p.y = pack2(m.r2.z, m.r2.w);
    p.z = pack2(m.r3.x, m.r3.y); p.w = pack2(m.r3.z, m.r3.w);
    br[1] = p;
    p.x = pack2(m.c0.x, m.c0.y); p.y = pack2(m.c0.z, m.c0.w);
    p.z = pack2(m.c1.x, m.c1.y); p.w = pack2(m.c1.z, m.c1.w);
    bi[0] = p;
    p.x = pack2(m.c2.x, m.c2.y); p.y = pack2(m.c2.z, m.c2.w);
    p.z = pack2(m.c3.x, m.c3.y); p.w = pack2(m.c3.z, m.c3.w);
    bi[1] = p;
  };

  // ---- WMMA burst over one 128x64(A) x 64x64(B) LDS tile (2 k-steps of 32) ----
  auto mma_tiles = [&](int buf) {
    const unsigned short* Ab = SM + A_OFF(buf);
    const unsigned short* BR = SM + BR_OFF(buf);
    const unsigned short* BI = SM + BI_OFF(buf);
#pragma unroll
    for (int kb = 0; kb < 2; ++kb) {
      bf16x16 afr[2];
#pragma unroll
      for (int mi = 0; mi < 2; ++mi) {
        const unsigned short* ap = Ab + (wm * 32 + mi * 16 + lr) * LDK + kb * 32 + kh * 16;
        FragCast f;
        f.q[0] = *(const U4*)(ap);        // K = base .. base+7
        f.q[1] = *(const U4*)(ap + 8);    // K = base+8 .. base+15
        afr[mi] = f.v;
      }
      bf16x16 bRf[2], bIf[2];
#pragma unroll
      for (int ni = 0; ni < 2; ++ni) {
        const int boff = (wn * 32 + ni * 16 + lr) * LDK + kb * 32 + kh * 16;
        FragCast f, g;
        f.q[0] = *(const U4*)(BR + boff);
        f.q[1] = *(const U4*)(BR + boff + 8);
        bRf[ni] = f.v;
        g.q[0] = *(const U4*)(BI + boff);
        g.q[1] = *(const U4*)(BI + boff + 8);
        bIf[ni] = g.v;
      }
#pragma unroll
      for (int mi = 0; mi < 2; ++mi)
#pragma unroll
        for (int ni = 0; ni < 2; ++ni) {
          accR[mi][ni] = __builtin_amdgcn_wmma_f32_16x16x32_bf16(
              false, afr[mi], false, bRf[ni], (short)0, accR[mi][ni], false, false);
          accI[mi][ni] = __builtin_amdgcn_wmma_f32_16x16x32_bf16(
              false, afr[mi], false, bIf[ni], (short)0, accI[mi][ni], false, false);
        }
    }
  };

  // ================= Phase 1: RBF contraction, K = 256 * 64, pipelined ======
  {
    MainRegs m0;
    load_main(0, m0);
    store_main(m0, 0);
  }
  __syncthreads();
  for (int i = 0; i < 256; ++i) {
    const int cur = i & 1;
    MainRegs nx;
    const bool have = (i + 1 < 256);
    if (have) load_main(i + 1, nx);    // global loads fly during the WMMA burst
    mma_tiles(cur);                    // fragment ds_loads issued before stores
    if (have) store_main(nx, cur ^ 1); // fill the other buffer
    __syncthreads();
  }

  // ================= Phase 2: SiLU GEMM tail, K = 512 (8 x 64) ==============
  for (int t = 0; t < 8; ++t) {
    const int half = (t >= 4);                 // 0: s_re rows, 1: s_im rows
    const int ibase = (t & 3) * 64;
    // A tile: silu(x) for k = ibase..ibase+63
    {
      const float* xp = (half ? x_im : x_re) + (size_t)(m_base + ar) * 256 + ibase + ah * 32;
      unsigned* arow = (unsigned*)(SM + A_OFF(0)) + ar * (LDK / 2) + ah * 16;
#pragma unroll
      for (int j2 = 0; j2 < 16; ++j2) {
        float x0 = xp[2 * j2], x1 = xp[2 * j2 + 1];
        float s0 = x0 / (1.0f + __expf(-x0));
        float s1 = x1 / (1.0f + __expf(-x1));
        arow[j2] = pack2(s0, s1);
      }
    }
    // B tiles (transposed gather): re-> [Wre;-Wim], im-> [Wim;Wre]
    {
      const float* srcR = half ? swi : swr;
      const float* srcI = half ? swr : swi;
      const float sgnR = half ? -1.0f : 1.0f;
      const int o = n_base + bo;
      unsigned short* brp = SM + BR_OFF(0) + bo * LDK + bq * 16;
      unsigned short* bip = SM + BI_OFF(0) + bo * LDK + bq * 16;
#pragma unroll
      for (int j = 0; j < 16; ++j) {
        const int irow = ibase + bq * 16 + j;
        brp[j] = f2bf(sgnR * srcR[irow * 256 + o]);
        bip[j] = f2bf(srcI[irow * 256 + o]);
      }
    }
    __syncthreads();
    mma_tiles(0);
    __syncthreads();
  }

  // ================= Epilogue: + bias, store (B, O, 2) fp32 =================
#pragma unroll
  for (int mi = 0; mi < 2; ++mi)
#pragma unroll
    for (int ni = 0; ni < 2; ++ni) {
      const int ocol_l = wn * 32 + ni * 16 + lr;
      const int ocol   = n_base + ocol_l;
      const float bR = biasR[ocol_l];
      const float bI = biasI[ocol_l];
      const int row0 = m_base + wm * 32 + mi * 16 + kh * 8;
#pragma unroll
      for (int rr = 0; rr < 8; ++rr) {
        const int brow = row0 + rr;
        float2 val = make_float2(accR[mi][ni][rr] + bR, accI[mi][ni][rr] + bI);
        *(float2*)(out + ((size_t)brow * 256 + (size_t)ocol) * 2) = val;
      }
    }
}

extern "C" void kernel_launch(void* const* d_in, const int* in_sizes, int n_in,
                              void* d_out, int out_size, void* d_ws, size_t ws_size,
                              hipStream_t stream) {
  const float* x_re = (const float*)d_in[0];
  const float* x_im = (const float*)d_in[1];
  const float* wR   = (const float*)d_in[2];
  const float* wC   = (const float*)d_in[3];
  const float* swr  = (const float*)d_in[4];
  const float* swi  = (const float*)d_in[5];
  const float* sbr  = (const float*)d_in[6];
  const float* sbi  = (const float*)d_in[7];
  float* out = (float*)d_out;

  const int Bn = in_sizes[0] / 256;            // 4096
  dim3 grid(Bn / TM, 256 / TN);                // (32, 4)
  cvkan_fused<<<grid, 256, SMEM_BYTES, stream>>>(x_re, x_im, wR, wC, swr, swi, sbr, sbi, out);
}
